// MultiHeadWKV_2319282340088
// MI455X (gfx1250) — matrix operand order repaired and checked by reading, not compiled, not run on hardware
//
#include <hip/hip_runtime.h>
#include <hip/hip_bf16.h>

#define DIMX   2048
#define NHEAD  32
#define DH     64
#define BSZ    4
#define TLEN   2048

typedef __bf16 bf16_t;
typedef __attribute__((ext_vector_type(16))) __bf16 v16bf;
typedef __attribute__((ext_vector_type(8)))  __bf16 v8bf;
typedef __attribute__((ext_vector_type(4)))  __bf16 v4bf;
typedef __attribute__((ext_vector_type(2)))  __bf16 v2bf;
typedef __attribute__((ext_vector_type(8)))  float  v8f;
typedef __attribute__((ext_vector_type(4)))  unsigned int v4u;
typedef __attribute__((ext_vector_type(8)))  int v8i;
typedef __attribute__((ext_vector_type(4)))  int v4i;

#if defined(__has_builtin)
#  if __has_builtin(__builtin_amdgcn_tensor_load_to_lds)
#    define HAVE_TDM 1
#  endif
#endif

// ---------------------------------------------------------------------------
// Kernel 1: WKV recurrence. One block of 256 threads per (b,h).
// thread = (column j = tid>>2, row-slice e = tid&3): owns 16 state rows of
// column j in VGPRs. Per step: 16 x (ds_load_b128 broadcast + 3 FMA), then a
// 4-lane shfl_xor reduction for y and s. 4x the step-level parallelism of a
// one-column-per-thread scheme; the t-loop stays inherently sequential.
// ---------------------------------------------------------------------------
__global__ __launch_bounds__(256) void wkv_kernel(
    const float* __restrict__ r, const float* __restrict__ w,
    const float* __restrict__ k, const float* __restrict__ v,
    const float* __restrict__ init_state, const float* __restrict__ u,
    float* __restrict__ xout, float* __restrict__ state_out)
{
    const int bh  = blockIdx.x;          // 0..127
    const int b   = bh / NHEAD;
    const int h   = bh % NHEAD;
    const int tid = threadIdx.x;
    const int j   = tid >> 2;            // column 0..63
    const int e   = tid & 3;             // row slice: rows 16e..16e+15

    __shared__ float4 sq[DH];            // (r_i, w_i, k_i, v_i)

    float S[16], uu[16];
    #pragma unroll
    for (int ii = 0; ii < 16; ++ii) {
        const int i = 16 * e + ii;
        S[ii]  = init_state[(h * DH + i) * DH + j];
        uu[ii] = u[h * DH + i];
    }

    const size_t base = (size_t)b * TLEN * DIMX + (size_t)h * DH;

    for (int t = 0; t < TLEN; ++t) {
        const size_t off = base + (size_t)t * DIMX;
        float4 p;
        if (tid < DH) {
            p.x = r[off + tid]; p.y = w[off + tid];
            p.z = k[off + tid]; p.w = v[off + tid];
        }
        __syncthreads();                 // previous step done reading sq
        if (tid < DH) sq[tid] = p;
        __syncthreads();

        const float vj = sq[j].w;
        float y = 0.f, s = 0.f;
        #pragma unroll
        for (int ii = 0; ii < 16; ++ii) {
            const float4 q = sq[16 * e + ii];
            y     = fmaf(q.x, S[ii], y);            // r_i * S[i][j]
            s     = fmaf(q.x * q.z, uu[ii], s);     // r_i*k_i*u_i
            S[ii] = fmaf(q.y, S[ii], q.z * vj);     // w_i*S + k_i*v_j
        }
        // reduce over the 4 row-slices (lanes tid^1, tid^2 are in-wave)
        y += __shfl_xor(y, 1, 32);  s += __shfl_xor(s, 1, 32);
        y += __shfl_xor(y, 2, 32);  s += __shfl_xor(s, 2, 32);
        if (e == 0) xout[off + j] = fmaf(s, vj, y);
    }

    #pragma unroll
    for (int ii = 0; ii < 16; ++ii) {
        const int i = 16 * e + ii;
        state_out[(((size_t)b * NHEAD + h) * DH + i) * DH + j] = S[ii];
    }
}

// ---------------------------------------------------------------------------
// Kernel 2: LayerNorm(64) + SiLU(g) gate, emit bf16 activations.
// ---------------------------------------------------------------------------
__global__ __launch_bounds__(256) void ln_gate_kernel(
    const float* __restrict__ x, const float* __restrict__ g,
    const float* __restrict__ ln_w, const float* __restrict__ ln_b,
    bf16_t* __restrict__ a)
{
    const int wave = blockIdx.x * 8 + (threadIdx.x >> 5);
    const int lane = threadIdx.x & 31;
    const size_t base = (size_t)wave * DH;
    const int c0 = lane * 2, c1 = c0 + 1;

    const float x0 = x[base + c0];
    const float x1 = x[base + c1];

    float sum = x0 + x1;
    float sq  = x0 * x0 + x1 * x1;
    #pragma unroll
    for (int o = 16; o > 0; o >>= 1) {
        sum += __shfl_xor(sum, o, 32);
        sq  += __shfl_xor(sq,  o, 32);
    }
    const float mu  = sum * (1.f / DH);
    const float var = sq * (1.f / DH) - mu * mu;
    const float inv = rsqrtf(var + 1e-5f);

    const float n0 = (x0 - mu) * inv * ln_w[c0] + ln_b[c0];
    const float n1 = (x1 - mu) * inv * ln_w[c1] + ln_b[c1];

    const float g0 = g[base + c0];
    const float g1 = g[base + c1];
    const float s0 = g0 / (1.f + __expf(-g0));
    const float s1 = g1 / (1.f + __expf(-g1));

    v2bf pv; pv[0] = (bf16_t)(s0 * n0); pv[1] = (bf16_t)(s1 * n1);
    *(v2bf*)(a + base + c0) = pv;
}

// ---------------------------------------------------------------------------
// Kernel 3: f32 -> bf16 weight conversion (o_w, 4M elements).
// ---------------------------------------------------------------------------
__global__ __launch_bounds__(256) void cvt_w_kernel(
    const float* __restrict__ w, bf16_t* __restrict__ o)
{
    const size_t i = ((size_t)blockIdx.x * 256 + threadIdx.x) * 4;
    const float4 f = *(const float4*)(w + i);
    v4bf pv;
    pv[0] = (bf16_t)f.x; pv[1] = (bf16_t)f.y;
    pv[2] = (bf16_t)f.z; pv[3] = (bf16_t)f.w;
    *(v4bf*)(o + i) = pv;
}

// ---------------------------------------------------------------------------
// Kernel 4: out = A @ B^T via v_wmma_f32_16x16x32_bf16.
// Block: 256 threads = 8 waves; block tile 128(M) x 128(N); wave tile 16x128.
// B tile (128 rows x 32 K, bf16) is staged in LDS once per block per K-step,
// double-buffered. Preferred path: Tensor Data Mover (tensor_load_to_lds,
// TENSORcnt-synchronized); fallback: cooperative copy. LDS rows are padded
// to 80 B (TDM pad_interval=16 DW, pad_amount=4 DW) so 32 B fragment reads
// stay 16 B-aligned with staggered banks.
// ---------------------------------------------------------------------------
#define LDSROW 80
#define BUFSZ  (128 * LDSROW)            // 10240 B per buffer

__global__ __launch_bounds__(256) void oproj_gemm_kernel(
    const bf16_t* __restrict__ A, const bf16_t* __restrict__ Bm,
    float* __restrict__ C)
{
    const int K = DIMX, N = DIMX;
    __shared__ char lbuf[2 * BUFSZ];

    const int tid    = threadIdx.x;
    const int waveId = tid >> 5;
    const int lane   = tid & 31;
    const int sl     = lane >> 4;        // K-half selector
    const int ln16   = lane & 15;

    const int tileN = blockIdx.x * 128;
    const int tileM = blockIdx.y * 128 + waveId * 16;

    const bf16_t* Arow = A + (size_t)(tileM + ln16) * K;

    v8f acc[8];
    #pragma unroll
    for (int nt = 0; nt < 8; ++nt)
        acc[nt] = (v8f){0.f, 0.f, 0.f, 0.f, 0.f, 0.f, 0.f, 0.f};

#ifdef HAVE_TDM
    typedef __attribute__((address_space(3))) char lds_char_t;
    const unsigned lds_base = (unsigned)(size_t)(lds_char_t*)&lbuf[0];
#endif

    // ---- stage helper: fill buffer `buf` with B[tileN..+127][kb..kb+31] ----
    auto stage = [&](int buf, int kb) {
#ifdef HAVE_TDM
        if (tid < 32) {                  // wave 0 issues one TDM descriptor
            const unsigned long long ga =
                (unsigned long long)(const void*)(Bm + (size_t)tileN * K + kb);
            v4u g0;
            g0[0] = 1u;                                       // count=1, user D#
            g0[1] = lds_base + (unsigned)(buf * BUFSZ);       // lds_addr
            g0[2] = (unsigned)ga;                             // global_addr lo
            g0[3] = (unsigned)((ga >> 32) & 0x1FFFFFFull) | (2u << 30); // hi|type=2
            v8i g1;
            g1[0] = 0x06D10000;          // data_size=2B, pad_en, intv=16DW, amt=4DW
            g1[1] = (int)((2048u & 0xFFFFu) << 16);           // tensor_dim0 lo
            g1[2] = (int)(((2048u >> 16)) | ((2048u & 0xFFFFu) << 16)); // d0hi|d1lo
            g1[3] = (int)(((2048u >> 16)) | (32u << 16));     // d1 hi | tile_dim0=32
            g1[4] = 128;                 // tile_dim1=128 rows, tile_dim2=0
            g1[5] = 2048;                // tensor_dim0_stride lo
            g1[6] = 0;                   // stride0 hi | stride1 lo
            g1[7] = 0;
            v4i gz = {0, 0, 0, 0};
#if __clang_major__ >= 23
            v8i gz8 = {0, 0, 0, 0, 0, 0, 0, 0};
            __builtin_amdgcn_tensor_load_to_lds(g0, g1, gz, gz, gz8, 0);
#else
            __builtin_amdgcn_tensor_load_to_lds(g0, g1, gz, gz, 0);
#endif
        }
#else
        // cooperative copy: 256 threads x 32 B = 128 rows x 64 B
        const int row = tid >> 1;
        const int hb  = tid & 1;
        const v8bf* src = (const v8bf*)(Bm + (size_t)(tileN + row) * K + kb + 16 * hb);
        v8bf* dst = (v8bf*)(lbuf + buf * BUFSZ + row * LDSROW + hb * 32);
        dst[0] = src[0];
        dst[1] = src[1];
#endif
    };

    stage(0, 0);                         // prologue fill
    int cur = 0;

    for (int kb = 0; kb < K; kb += 32) {
        const bool more = (kb + 32) < K;
        if (more) stage(cur ^ 1, kb + 32);
#ifdef HAVE_TDM
        if (tid < 32) {
            if (more) __builtin_amdgcn_s_wait_tensorcnt(1);  // older load done
            else      __builtin_amdgcn_s_wait_tensorcnt(0);
        }
#endif
        __syncthreads();                 // buf[cur] ready for everyone

        __builtin_prefetch(Arow + kb + 64, 0, 1);            // next A tile

        // A fragment (16-bit A 16x32): elems 0..15 = K kb+16*sl.. contiguous
        const v8bf alo = *(const v8bf*)(Arow + kb + 16 * sl);
        const v8bf ahi = *(const v8bf*)(Arow + kb + 16 * sl + 8);
        v16bf af;
        #pragma unroll
        for (int ee = 0; ee < 8; ++ee) { af[ee] = alo[ee]; af[ee + 8] = ahi[ee]; }

        const char* bufp = lbuf + cur * BUFSZ;
        #pragma unroll
        for (int nt = 0; nt < 8; ++nt) {
            const int rr = nt * 16 + ln16;                   // B row in tile
            const v16bf bfr = *(const v16bf*)(bufp + rr * LDSROW + sl * 32);
            acc[nt] = __builtin_amdgcn_wmma_f32_16x16x32_bf16(
                false, af, false, bfr, (short)0, acc[nt], false, false);
        }

        __syncthreads();                 // reads done before buf[cur] restaged
        cur ^= 1;
    }

    // D layout: lane (ln16 + 16*sl), VGPR q -> row tileM + q + 8*sl
    #pragma unroll
    for (int nt = 0; nt < 8; ++nt) {
        const int col = tileN + nt * 16 + ln16;
        #pragma unroll
        for (int q = 0; q < 8; ++q)
            C[(size_t)(tileM + q + 8 * sl) * N + col] = acc[nt][q];
    }
}

// ---------------------------------------------------------------------------
extern "C" void kernel_launch(void* const* d_in, const int* in_sizes, int n_in,
                              void* d_out, int out_size, void* d_ws, size_t ws_size,
                              hipStream_t stream)
{
    const float* r          = (const float*)d_in[0];
    const float* w          = (const float*)d_in[1];
    const float* k          = (const float*)d_in[2];
    const float* v          = (const float*)d_in[3];
    const float* g          = (const float*)d_in[4];
    const float* init_state = (const float*)d_in[5];
    const float* u          = (const float*)d_in[6];
    const float* ln_w       = (const float*)d_in[7];
    const float* ln_b       = (const float*)d_in[8];
    const float* o_w        = (const float*)d_in[9];

    float* out       = (float*)d_out;                       // [B,T,DIM]
    float* state_out = out + (size_t)BSZ * TLEN * DIMX;     // [B,H,DH,DH]

    char*   ws  = (char*)d_ws;
    float*  x   = (float*)ws;                                             // 64 MB f32
    bf16_t* a   = (bf16_t*)(ws + (size_t)BSZ * TLEN * DIMX * 4);          // 32 MB bf16
    bf16_t* owb = (bf16_t*)(ws + (size_t)BSZ * TLEN * DIMX * 4
                               + (size_t)BSZ * TLEN * DIMX * 2);          //  8 MB bf16

    wkv_kernel<<<BSZ * NHEAD, 256, 0, stream>>>(r, w, k, v, init_state, u, x, state_out);

    ln_gate_kernel<<<(BSZ * TLEN * NHEAD) / 8, 256, 0, stream>>>(x, g, ln_w, ln_b, a);

    cvt_w_kernel<<<(DIMX * DIMX / 4) / 256, 256, 0, stream>>>(o_w, owb);

    oproj_gemm_kernel<<<dim3(DIMX / 128, (BSZ * TLEN) / 128), 256, 0, stream>>>(a, owb, out);
}
